// Sggnn_gcn_48000554500541
// MI455X (gfx1250) — matrix operand order repaired and checked by reading, not compiled, not run on hardware
//
#include <hip/hip_runtime.h>
#include <hip/hip_bf16.h>

// ---------------------------------------------------------------------------
// CDNA5 (gfx1250, wave32) implementation of the SGGNN-GCN reference.
// All matmuls run on v_wmma_f32_16x16x32_bf16 (bf16 operands, f32 accum).
// Block tile 128x128, wave tile 32x64 (2x4 WMMA), K-step 32.
// bf16 A-operand staging uses GLOBAL_LOAD_ASYNC_TO_LDS_B128 (ASYNCcnt).
// ---------------------------------------------------------------------------

typedef __attribute__((ext_vector_type(16))) __bf16        v16bf;
typedef __attribute__((ext_vector_type(8)))  float         v8f;
typedef __attribute__((ext_vector_type(8)))  unsigned int  v8u;

union Frag { v8u u; v16bf b; };

#define BN_SCALE_F 0.9999950000374997f  // 1/sqrt(1+1e-5)

#if __has_builtin(__builtin_amdgcn_global_load_async_to_lds_b128) && \
    __has_builtin(__builtin_amdgcn_s_wait_asynccnt)
#define HAVE_ASYNC_LDS 1
typedef int v4i_ __attribute__((vector_size(16)));   // matches builtin param
#else
#define HAVE_ASYNC_LDS 0
#endif

__device__ __forceinline__ unsigned short f2bf(float f) {
  unsigned int u = __builtin_bit_cast(unsigned int, f);
  u += 0x7FFFu + ((u >> 16) & 1u);           // round-to-nearest-even
  return (unsigned short)(u >> 16);
}
__device__ __forceinline__ float bf2f(unsigned short h) {
  unsigned int u = ((unsigned int)h) << 16;
  return __builtin_bit_cast(float, u);
}
__device__ __forceinline__ unsigned int pack2(float a, float b) {
  return (unsigned int)f2bf(a) | ((unsigned int)f2bf(b) << 16);
}

// ---------------------------------------------------------------------------
// WMMA GEMM: C[M,N] (bf16) = epi( A[M,K] @ B[K,N] )
// 256 threads = 8 waves (4 M-waves x 2 N-waves).
// Block tile 128x128, K-step 32. Wave tile 32x64 = 2x4 of 16x16x32 WMMA.
// A operand f32 or bf16 (converted during LDS staging), same for B.
// EPI=1: out = lrelu( g*BN_SCALE*(acc + lb) + be ), per output column.
// LDS row stride = 40 halves (20 dwords): 16B-aligned rows AND rows map to
// distinct banks for the fragment gathers.
// ---------------------------------------------------------------------------
template<int EPI, int AF32, int BF32>
__global__ __launch_bounds__(256) void gemm_kernel(
    const void* __restrict__ Aptr, const void* __restrict__ Bptr,
    unsigned short* __restrict__ Cptr,
    int M, int N, int K,
    long long sAz, long long sBz, long long sCz,
    const float* __restrict__ gvec, const float* __restrict__ lbias,
    const float* __restrict__ bevec)
{
  __shared__ unsigned int lds[(128 * 40 + 128 * 40) / 2];  // 20480 bytes
  unsigned int*   ldsA32 = lds;                  // A: 128 rows x 20 dwords
  unsigned int*   ldsB32 = lds + 128 * 20;       // B^T: 128 cols x 20 dwords
  unsigned short* ldsBh  = (unsigned short*)ldsB32;

  const int tid   = threadIdx.x;
  const int lane  = tid & 31;
  const int wave  = tid >> 5;
  const int l16   = lane & 15;
  const int hi    = lane >> 4;                   // half-wave select
  const int waveM = (wave & 3) * 32;
  const int waveN = (wave >> 2) * 64;

  const int rowBlock = blockIdx.y * 128;
  const int colBlock = blockIdx.x * 128;
  const int z        = blockIdx.z;

  const float*          Af = (const float*)Aptr          + (size_t)z * sAz;
  const unsigned short* Ah = (const unsigned short*)Aptr + (size_t)z * sAz;
  const float*          Bf = (const float*)Bptr          + (size_t)z * sBz;
  const unsigned short* Bh = (const unsigned short*)Bptr + (size_t)z * sBz;
  unsigned short*       C  = Cptr                        + (size_t)z * sCz;

  v8f acc[2][4] = {};

  for (int k0 = 0; k0 < K; k0 += 32) {
    __syncthreads();
    // ---- stage A tile (128 x 32) -> LDS row-major -------------------------
    if (AF32) {
      const float* Ab = Af + (size_t)rowBlock * K + k0;
#pragma unroll
      for (int p = 0; p < 8; ++p) {
        int lin = p * 256 + tid;                 // 2048 float-pairs
        int r = lin >> 4, c2 = lin & 15;
        const float* s = Ab + (size_t)r * K + c2 * 2;
        ldsA32[r * 20 + c2] = pack2(s[0], s[1]);
      }
      if (k0 + 32 < K)                           // global_prefetch_b8 flavor
        __builtin_prefetch(Ab + (size_t)(tid >> 1) * K + 32 + (tid & 1) * 16, 0, 1);
    } else {
#if HAVE_ASYNC_LDS
      // 8 KiB tile via 512 async b128 transfers (ASYNCcnt-tracked)
      const unsigned short* Ab = Ah + (size_t)rowBlock * K + k0;
#pragma unroll
      for (int p = 0; p < 2; ++p) {
        int lin = p * 256 + tid;                 // 512 chunks of 8 halves
        int r = lin >> 2, c8 = lin & 3;
        const unsigned short* gp = Ab + (size_t)r * K + c8 * 8;
        char* lp = (char*)ldsA32 + r * 80 + c8 * 16;
        __builtin_amdgcn_global_load_async_to_lds_b128(
            (__attribute__((address_space(1))) v4i_*)gp,
            (__attribute__((address_space(3))) v4i_*)lp, 0, 0);
      }
      __builtin_amdgcn_s_wait_asynccnt(0);
#else
      const unsigned short* Ab = Ah + (size_t)rowBlock * K + k0;
#pragma unroll
      for (int p = 0; p < 4; ++p) {
        int lin = p * 256 + tid;                 // 1024 x uint2 (4 halves)
        int r = lin >> 3, c = lin & 7;
        uint2 v = *(const uint2*)(Ab + (size_t)r * K + c * 4);
        ldsA32[r * 20 + c * 2]     = v.x;
        ldsA32[r * 20 + c * 2 + 1] = v.y;
      }
#endif
    }
    // ---- stage B tile (32 x 128) transposed -> LDS [n][k] -----------------
#pragma unroll
    for (int p = 0; p < 8; ++p) {
      int lin = p * 256 + tid;                   // 2048 n-pairs
      int kk = lin >> 6, n2 = (lin & 63) * 2;
      if (BF32) {
        const float* s = Bf + (size_t)(k0 + kk) * N + colBlock + n2;
        float2 v = *(const float2*)s;
        ldsBh[n2 * 40 + kk]       = f2bf(v.x);
        ldsBh[(n2 + 1) * 40 + kk] = f2bf(v.y);
      } else {
        unsigned int u = *(const unsigned int*)(Bh + (size_t)(k0 + kk) * N + colBlock + n2);
        ldsBh[n2 * 40 + kk]       = (unsigned short)u;
        ldsBh[(n2 + 1) * 40 + kk] = (unsigned short)(u >> 16);
      }
    }
    __syncthreads();

    // ---- gather fragments per ISA 16-bit A(16x32)/B(32x16) layouts --------
    Frag afr[2], bfr[4];
#pragma unroll
    for (int mi = 0; mi < 2; ++mi) {
      int r = waveM + mi * 16 + l16;
      const unsigned int* p = ldsA32 + r * 20 + hi * 4;
#pragma unroll
      for (int v = 0; v < 4; ++v) { afr[mi].u[v] = p[v]; afr[mi].u[4 + v] = p[8 + v]; }
    }
#pragma unroll
    for (int ni = 0; ni < 4; ++ni) {
      int c = waveN + ni * 16 + l16;
      const unsigned int* q = ldsB32 + c * 20 + hi * 4;
#pragma unroll
      for (int v = 0; v < 4; ++v) { bfr[ni].u[v] = q[v]; bfr[ni].u[4 + v] = q[8 + v]; }
    }
#pragma unroll
    for (int mi = 0; mi < 2; ++mi)
#pragma unroll
      for (int ni = 0; ni < 4; ++ni)
        acc[mi][ni] = __builtin_amdgcn_wmma_f32_16x16x32_bf16(
            false, afr[mi].b, false, bfr[ni].b, (short)0, acc[mi][ni],
            false, false);
  }

  // ---- epilogue: fused BN affine + LeakyReLU, store bf16 ------------------
#pragma unroll
  for (int mi = 0; mi < 2; ++mi) {
#pragma unroll
    for (int ni = 0; ni < 4; ++ni) {
      int col = colBlock + waveN + ni * 16 + l16;
      float a = 1.0f, bb = 0.0f;
      if (EPI) {
        a  = gvec[col] * BN_SCALE_F;
        bb = a * lbias[col] + bevec[col];
      }
#pragma unroll
      for (int e = 0; e < 8; ++e) {
        int row = rowBlock + waveM + mi * 16 + e + 8 * hi;  // C-matrix layout
        float v = acc[mi][ni][e];
        if (EPI) {
          v = a * v + bb;
          v = v > 0.0f ? v : 0.1f * v;
        }
        C[(size_t)row * N + col] = f2bf(v);
      }
    }
  }
}

// ---------------------------------------------------------------------------
// dinv[i] = (1 + rowsum(w[i,:]))^-0.5
// ---------------------------------------------------------------------------
__global__ void rowsum_kernel(const float* __restrict__ w,
                              float* __restrict__ dinv, int N) {
  __shared__ float red[256];
  int row = blockIdx.x;
  const float* wr = w + (size_t)row * N;
  float s = 0.0f;
  for (int j = threadIdx.x; j < N; j += 256) s += wr[j];
  red[threadIdx.x] = s;
  __syncthreads();
  for (int off = 128; off > 0; off >>= 1) {
    if (threadIdx.x < off) red[threadIdx.x] += red[threadIdx.x + off];
    __syncthreads();
  }
  if (threadIdx.x == 0) dinv[row] = 1.0f / sqrtf(red[0] + 1.0f);
}

// A[m,n] = (w[m,n] + delta(m,n)) * dinv[m] * dinv[n]  ( = w_norm^T )
__global__ void build_graph_kernel(const float* __restrict__ w,
                                   const float* __restrict__ dinv,
                                   unsigned short* __restrict__ Abf, int N) {
  int idx = blockIdx.x * 256 + threadIdx.x;
  int m = idx / N, n = idx - m * N;
  float v = w[idx] + (m == n ? 1.0f : 0.0f);
  Abf[idx] = f2bf(v * dinv[m] * dinv[n]);
}

// result[r] = dot(Y[r,:], cls_W) + cls_b  -- one wave32 per row
__global__ void cls_kernel(const unsigned short* __restrict__ Y,
                           const float* __restrict__ cw,
                           const float* __restrict__ cb,
                           float* __restrict__ out, int H) {
  int wave = threadIdx.x >> 5, lane = threadIdx.x & 31;
  int row = blockIdx.x * 8 + wave;
  const unsigned short* y = Y + (size_t)row * H;
  float s = 0.0f;
  for (int h = lane; h < H; h += 32) s += bf2f(y[h]) * cw[h];
#pragma unroll
  for (int off = 16; off > 0; off >>= 1) s += __shfl_xor(s, off, 32);
  if (lane == 0) out[row] = s + cb[0];
}

__global__ void copy_i32_kernel(const int* __restrict__ src,
                                int* __restrict__ dst, int n) {
  int i = blockIdx.x * 256 + threadIdx.x;
  if (i < n) dst[i] = src[i];
}

// ---------------------------------------------------------------------------
extern "C" void kernel_launch(void* const* d_in, const int* in_sizes, int n_in,
                              void* d_out, int out_size, void* d_ws, size_t ws_size,
                              hipStream_t stream) {
  constexpr int Bb = 16, G = 128, F = 1024, H = 512;
  constexpr int N    = Bb * G;        // 2048 graph nodes
  constexpr int ROWS = Bb * Bb * G;   // 32768 total rows

  const float* d     = (const float*)d_in[0];
  const float* w     = (const float*)d_in[1];
  const int*   label = (const int*)  d_in[2];
  const float* W1 = (const float*)d_in[3];
  const float* b1 = (const float*)d_in[4];
  const float* g1 = (const float*)d_in[5];
  const float* e1 = (const float*)d_in[6];
  const float* W2 = (const float*)d_in[7];
  const float* b2 = (const float*)d_in[8];
  const float* g2 = (const float*)d_in[9];
  const float* e2 = (const float*)d_in[10];
  const float* Wf = (const float*)d_in[11];
  const float* bf = (const float*)d_in[12];
  const float* gf = (const float*)d_in[13];
  const float* ef = (const float*)d_in[14];
  const float* Wc = (const float*)d_in[15];
  const float* bc = (const float*)d_in[16];

  // workspace layout (needs ~136.4 MiB):
  //   dinv 8KB | Abf 8MiB (N*N bf16) | buf0 64MiB | buf1 64MiB
  char* ws = (char*)d_ws;
  float*          dinv = (float*)ws;
  unsigned short* Abf  = (unsigned short*)(ws + (8 << 10));
  unsigned short* buf0 = (unsigned short*)(ws + (8 << 10) + (size_t)N * N * 2);
  unsigned short* buf1 = buf0 + (size_t)ROWS * F;

  rowsum_kernel<<<N, 256, 0, stream>>>(w, dinv, N);
  build_graph_kernel<<<(N * N) / 256, 256, 0, stream>>>(w, dinv, Abf, N);

  // X1 = lrelu(bn1(d @ W1 + b1))            [32768 x 1024] -> buf0
  gemm_kernel<1, 1, 1><<<dim3(F / 128, ROWS / 128, 1), 256, 0, stream>>>(
      d, W1, buf0, ROWS, F, F, 0, 0, 0, g1, b1, e1);
  // t  = lrelu(bn2(X1 @ W2 + b2))           [32768 x 1024] -> buf1
  gemm_kernel<1, 0, 1><<<dim3(F / 128, ROWS / 128, 1), 256, 0, stream>>>(
      buf0, W2, buf1, ROWS, F, F, 0, 0, 0, g2, b2, e2);
  // dp[b] = A @ t[b]                        16 x [2048 x 1024] -> buf0
  gemm_kernel<0, 0, 0><<<dim3(F / 128, N / 128, Bb), 256, 0, stream>>>(
      Abf, buf1, buf0, N, F, N, 0, (long long)N * F, (long long)N * F,
      nullptr, nullptr, nullptr);
  // Y  = lrelu(bn3(dp @ Wf + bf))           [32768 x 512] -> buf1
  gemm_kernel<1, 0, 1><<<dim3(H / 128, ROWS / 128, 1), 256, 0, stream>>>(
      buf0, Wf, buf1, ROWS, H, F, 0, 0, 0, gf, bf, ef);
  // result = Y @ Wc + bc                    [32768] -> d_out (f32)
  cls_kernel<<<ROWS / 8, 256, 0, stream>>>(buf1, Wc, bc, (float*)d_out, H);
  // tuple output #2: label passthrough (int32 bits)
  copy_i32_kernel<<<ROWS / 256, 256, 0, stream>>>(label, (int*)d_out + ROWS, ROWS);

  (void)in_sizes; (void)n_in; (void)out_size; (void)ws_size;
}
